// FPAC_Layer_46677704573558
// MI455X (gfx1250) — matrix operand order repaired and checked by reading, not compiled, not run on hardware
//
#include <hip/hip_runtime.h>
#include <hip/hip_bf16.h>
#include <math.h>

typedef __attribute__((ext_vector_type(16))) _Float16 v16h;
typedef __attribute__((ext_vector_type(8)))  float    v8f;
typedef __attribute__((ext_vector_type(8)))  unsigned v8u;

#define DEVINL __device__ __forceinline__

constexpr int cB = 8, cN = 512, cCIN = 64, cMAXN = 16, cNF = 32, cMID = 16;
constexpr int cBN = cB * cN;               // 4096 points
constexpr int cM  = cBN * cMAXN;           // 65536 sl rows (w/o fp rows)
constexpr int cRT = (cM + cNF) * cNF;      // 2098176 diff rows
constexpr float cEPS = 1e-3f;

// ---------------- workspace layout (bytes) ----------------
constexpr size_t OFF_FP    = 0;                          // 96 f32 rotated framepoints
constexpr size_t OFF_IDX   = 512;                        // 65536 i32 knn indices
constexpr size_t OFF_REL   = OFF_IDX  + 65536ull*4;      // 196608 f32 rel xyz
constexpr size_t OFF_STATS = OFF_REL  + 196608ull*4;     // 8192 f32 stats/affine
constexpr size_t OFF_W1H   = OFF_STATS+ 8192ull*4;       // 1024 f16  m1_w1
constexpr size_t OFF_WFH   = OFF_W1H  + 1024ull*2;       // 512 f16   final w [32,16]
constexpr size_t OFF_MRW1H = OFF_WFH  + 512ull*2;        // 32768 f16 mr_w1
constexpr size_t OFF_MRW0H = OFF_MRW1H+ 32768ull*2;      // 524288 f16 mr_w0
constexpr size_t OFF_M2T1  = OFF_MRW0H+ 524288ull*2;     // 16384 f32 m2 layer0 act
constexpr size_t OFF_HPRE  = OFF_M2T1 + 16384ull*4;      // cRT f32  h pre-BN
constexpr size_t OFF_WMAIN = OFF_HPRE + (size_t)cRT*4;   // 1048576 f32 agg
constexpr size_t OFF_FH    = OFF_WMAIN+ 1048576ull*4;    // 4194304 f16 einsum out
constexpr size_t OFF_R1    = OFF_FH   + 4194304ull*2;    // 2097152 f32 mr l0 pre-BN
constexpr size_t OFF_R1H   = OFF_R1   + 2097152ull*4;    // 2097152 f16 mr l0 post-BN
constexpr size_t OFF_R2    = OFF_R1H  + 2097152ull*2;    // 262144 f32 mr l1 pre-BN

// stats region sub-offsets (f32 units): [sum..][sq..] then [scale..][shift..]
constexpr int ST_BN1 = 0,   ST_BN1A = 64;
constexpr int ST_BN2 = 128, ST_BN2A = 192;
constexpr int ST_BN3 = 256, ST_BN3A = 260;
constexpr int ST_M2B0 = 512;                  // 512 sum + 512 sq
constexpr int ST_MRB0 = 2560, ST_MRB0A = 3584;
constexpr int ST_MRB1 = 4608, ST_MRB1A = 4736;

// ---------------- device helpers ----------------
DEVINL unsigned rotl32(unsigned x, int r){ return (x<<r)|(x>>(32-r)); }

// threefry2x32 reproducing jax.random.uniform(jax.random.key(42), ())
DEVINL float jax_uniform42(){
  const unsigned k0 = 0u, k1 = 42u;
  const unsigned ks2 = 0x1BD11BDAu ^ k0 ^ k1;
  unsigned x0 = k0, x1 = k1;  // count = [0,0] then += keys
  const int R0[4] = {13,15,26,6}, R1[4] = {17,29,16,24};
  #pragma unroll
  for (int i=0;i<4;i++){ x0 += x1; x1 = rotl32(x1,R0[i]); x1 ^= x0; }
  x0 += k1;  x1 += ks2 + 1u;
  #pragma unroll
  for (int i=0;i<4;i++){ x0 += x1; x1 = rotl32(x1,R1[i]); x1 ^= x0; }
  x0 += ks2; x1 += k0 + 2u;
  #pragma unroll
  for (int i=0;i<4;i++){ x0 += x1; x1 = rotl32(x1,R0[i]); x1 ^= x0; }
  x0 += k0;  x1 += k1 + 3u;
  #pragma unroll
  for (int i=0;i<4;i++){ x0 += x1; x1 = rotl32(x1,R1[i]); x1 ^= x0; }
  x0 += k1;  x1 += ks2 + 4u;
  #pragma unroll
  for (int i=0;i<4;i++){ x0 += x1; x1 = rotl32(x1,R0[i]); x1 ^= x0; }
  x0 += ks2; x1 += k0 + 5u;
  unsigned u = (x0 >> 9) | 0x3F800000u;
  return __uint_as_float(u) - 1.0f;
}

// mish(v) = v*tanh(softplus(v)); tanh(ln(1+u)) == (u^2+2u)/(u^2+2u+2), u=e^v.
// One v_exp_f32 + one v_rcp_f32, branchless clamp for large v.
DEVINL float mishf(float v){
  float u = __expf(v);
  float n = u * (u + 2.f);
  float r = n * __builtin_amdgcn_rcpf(n + 2.f);
  return v * ((v > 20.f) ? 1.f : r);
}

DEVINL float wave_sum2(float v){
  #pragma unroll
  for (int o=16;o>0;o>>=1) v += __shfl_xor(v, o, 32);
  return v;
}

DEVINL v8f wmma_f16(v16h a, v16h b, v8f c){
  return __builtin_amdgcn_wmma_f32_16x16x32_f16(false, a, false, b, (short)0, c, false, false);
}

// A fragment (16x32 f16) from a row-major f16 matrix, per-lane row, K window k0..k0+31
DEVINL v16h frag_a16(const _Float16* __restrict__ base, int ldk, int row, int k0, int lane){
  int hs = (lane >= 16) ? 8 : 0;
  const _Float16* rp = base + (size_t)row*ldk + k0;
  v8u au;
  #pragma unroll
  for (int p=0;p<8;p++){
    int kk = ((p<4)?0:16) + hs + ((p&3)<<1);
    au[p] = *(const unsigned*)(rp + kk);   // 2 consecutive halfs
  }
  return __builtin_bit_cast(v16h, au);
}

// B fragment (32x16 f16) from row-major [32][ldn] f16 matrix, N columns n0..n0+15
DEVINL v16h frag_b16(const _Float16* __restrict__ b, int ldn, int n0, int lane){
  int nn = n0 + (lane & 15);
  int kb = (lane >= 16) ? 8 : 0;
  v16h r;
  #pragma unroll
  for (int h=0;h<16;h++) r[h] = b[(size_t)(kb+(h&7)+((h>>3)<<4))*ldn + nn];
  return r;
}

DEVINL void load_diff(int r, const float* __restrict__ rel, const float* sfp,
                      float& dx, float& dy, float& dz){
  int i = r >> 5, j = r & 31;
  if (i < cM){
    dx = rel[i*3+0]-sfp[j*3+0]; dy = rel[i*3+1]-sfp[j*3+1]; dz = rel[i*3+2]-sfp[j*3+2];
  } else {
    int ii = i - cM;
    dx = sfp[ii*3+0]-sfp[j*3+0]; dy = sfp[ii*3+1]-sfp[j*3+1]; dz = sfp[ii*3+2]-sfp[j*3+2];
  }
}

// A fragment of BN1-normalized m1 layer-1 activations (recomputed on the fly)
DEVINL v16h frag_a1(int r, int lane, const float* __restrict__ rel,
                    const float* sfp, const float* sw, const float* sb, const float* sa){
  float dx, dy, dz;
  load_diff(r, rel, sfp, dx, dy, dz);
  int hs = (lane >= 16) ? 8 : 0;
  v16h a;
  #pragma unroll
  for (int h=0;h<16;h++){
    int p = h >> 1;
    int k = ((p<4)?0:16) + hs + ((p&3)<<1) + (h&1);
    float v = mishf(sb[k] + dx*sw[k] + dy*sw[32+k] + dz*sw[64+k]);
    a[h] = (_Float16)fmaf(v, sa[k], sa[32+k]);
  }
  return a;
}

// ---------------- kernels ----------------
__global__ void k_prep(const float* __restrict__ fpts, const float* __restrict__ m1w1,
                       const float* __restrict__ mrw1, float* __restrict__ fp,
                       _Float16* __restrict__ w1h, _Float16* __restrict__ mrw1h){
  int tid = threadIdx.x;
  if (tid < cNF){
    float ang = jax_uniform42() * 6.283185307179586477f;
    float cs = cosf(ang), sn = sinf(ang);
    float fx = fpts[tid*3], fy = fpts[tid*3+1], fz = fpts[tid*3+2];
    fp[tid*3+0] = fx*cs - fz*sn;
    fp[tid*3+1] = fy;
    fp[tid*3+2] = fx*sn + fz*cs;
  }
  for (int i=tid; i<1024;  i+=256) w1h[i]   = (_Float16)m1w1[i];
  for (int i=tid; i<32768; i+=256) mrw1h[i] = (_Float16)mrw1[i];
}

__global__ void k_conv_f2h(const float* __restrict__ in, _Float16* __restrict__ out, int n){
  int i = blockIdx.x*256 + threadIdx.x;
  if (i < n) out[i] = (_Float16)in[i];
}

// kNN: one thread per query point, top-16 by feature distance (ascending, stable)
__global__ void k_knn(const float* __restrict__ x, int* __restrict__ out){
  int t = blockIdx.x*256 + threadIdx.x;
  int b = t >> 9, n = t & 511;
  const float4* xb = (const float4*)(x + (size_t)b*cN*cCIN);
  float4 q[16];
  #pragma unroll
  for (int c=0;c<16;c++) q[c] = xb[n*16+c];
  float bd[16]; int bi[16];
  #pragma unroll
  for (int k=0;k<16;k++){ bd[k]=3.4e38f; bi[k]=0; }
  for (int m=0;m<512;m++){
    __builtin_prefetch(xb + ((m+4)&511)*16, 0, 0);
    float d = 0.f;
    #pragma unroll
    for (int c=0;c<16;c++){
      float4 v = xb[m*16+c];
      float a0=q[c].x-v.x, a1=q[c].y-v.y, a2=q[c].z-v.z, a3=q[c].w-v.w;
      d = fmaf(a0,a0,d); d = fmaf(a1,a1,d); d = fmaf(a2,a2,d); d = fmaf(a3,a3,d);
    }
    if (d < bd[15]){
      int pos = 15;
      #pragma unroll
      for (int k=14;k>=0;k--) if (d < bd[k]) pos = k;
      #pragma unroll
      for (int k=15;k>0;k--) if (k > pos){ bd[k]=bd[k-1]; bi[k]=bi[k-1]; }
      #pragma unroll
      for (int k=0;k<16;k++) if (k == pos){ bd[k]=d; bi[k]=m; }
    }
  }
  #pragma unroll
  for (int k=0;k<16;k++) out[t*16+k] = bi[k];
}

__global__ void k_rel(const float* __restrict__ xyz, const int* __restrict__ idx,
                      float* __restrict__ rel){
  int r = blockIdx.x*256 + threadIdx.x;      // 0..cM-1
  int t = r >> 4, b = t >> 9;
  int i0 = idx[t*16], ij = idx[r];
  const float* base = xyz + (size_t)b*cN*3;
  rel[r*3+0] = base[ij*3+0] - base[i0*3+0];
  rel[r*3+1] = base[ij*3+1] - base[i0*3+1];
  rel[r*3+2] = base[ij*3+2] - base[i0*3+2];
}

// m2 layer 0: [32,4096] @ [4096,512] + mish + stats
__global__ void k_m2a(const float* __restrict__ fpw, const float* __restrict__ w0,
                      const float* __restrict__ b0, float* __restrict__ t1,
                      float* __restrict__ stats){
  int t = blockIdx.x*256 + threadIdx.x;      // 16384 = 32x512
  int r = t >> 9, c = t & 511;
  const float* a = fpw + (size_t)r*4096;
  float acc = b0[c];
  for (int k=0;k<4096;k++) acc = fmaf(a[k], w0[(size_t)k*512 + c], acc);
  float v = mishf(acc);
  t1[r*512 + c] = v;
  atomicAdd(&stats[c], v);
  atomicAdd(&stats[512 + c], v*v);
}

// m2: finalize BN0, layer 1 (512->16), in-block BN1 -> final w [32,16] f16
__global__ void k_m2b(const float* __restrict__ t1, const float* __restrict__ stats,
                      const float* __restrict__ g0, const float* __restrict__ be0,
                      const float* __restrict__ w1, const float* __restrict__ b1,
                      const float* __restrict__ g1, const float* __restrict__ be1,
                      _Float16* __restrict__ wfh){
  __shared__ float s0[512], t0[512], o1[512], red[32];
  int tid = threadIdx.x;
  {
    float mu  = stats[tid] * (1.f/32.f);
    float var = stats[512+tid]*(1.f/32.f) - mu*mu;
    float s = g0[tid] * rsqrtf(var + cEPS);
    s0[tid] = s; t0[tid] = be0[tid] - mu*s;
  }
  __syncthreads();
  int r = tid >> 4, d = tid & 15;
  float acc = b1[d];
  for (int c=0;c<512;c++)
    acc = fmaf(fmaf(t1[r*512+c], s0[c], t0[c]), w1[c*16+d], acc);
  float v = mishf(acc);
  o1[tid] = v;
  __syncthreads();
  if (tid < 16){
    float s=0.f, q=0.f;
    for (int rr=0;rr<32;rr++){ float u = o1[rr*16+tid]; s+=u; q+=u*u; }
    float mu = s*(1.f/32.f), var = q*(1.f/32.f) - mu*mu;
    float sc = g1[tid]*rsqrtf(var + cEPS);
    red[tid] = sc; red[16+tid] = be1[tid] - mu*sc;
  }
  __syncthreads();
  wfh[tid] = (_Float16)fmaf(v, red[d], red[16+d]);
}

// m1 layer 1 stats pass (3->32), column sum/sumsq via shuffle + LDS + global atomics
__global__ void k_h1(const float* __restrict__ rel, const float* __restrict__ fp,
                     const float* __restrict__ w0, const float* __restrict__ b0,
                     float* __restrict__ stats){
  __shared__ float sfp[96], sw[96], sb[32], ls[64];
  int tid = threadIdx.x, lane = tid & 31;
  if (tid < 96){ sfp[tid]=fp[tid]; sw[tid]=w0[tid]; }
  if (tid < 32) sb[tid]=b0[tid];
  if (tid < 64) ls[tid]=0.f;
  __syncthreads();
  int r = blockIdx.x*256 + tid;
  float dx, dy, dz;
  load_diff(r, rel, sfp, dx, dy, dz);
  #pragma unroll
  for (int c=0;c<32;c++){
    float v = mishf(sb[c] + dx*sw[c] + dy*sw[32+c] + dz*sw[64+c]);
    float s = wave_sum2(v);
    float q = wave_sum2(v*v);
    if (lane == 0){ atomicAdd(&ls[c], s); atomicAdd(&ls[32+c], q); }
  }
  __syncthreads();
  if (tid < 64) atomicAdd(&stats[tid], ls[tid]);
}

// generic BN finalize: affine scale/shift from raw sums
__global__ void k_bnfin(const float* __restrict__ stats, float* __restrict__ aff,
                        const float* __restrict__ g, const float* __restrict__ be,
                        float cnt, int n){
  int c = blockIdx.x*blockDim.x + threadIdx.x;
  if (c >= n) return;
  float mu  = stats[c] / cnt;
  float var = stats[n+c] / cnt - mu*mu;
  float s = g[c] * rsqrtf(var + cEPS);
  aff[c] = s; aff[n+c] = be[c] - mu*s;
}

// m1 layer 2 stats pass: WMMA 16x16x32, per-column sums of mish(a1n@W1+b1)
__global__ void k_h2(const float* __restrict__ rel, const float* __restrict__ fp,
                     const float* __restrict__ w0, const float* __restrict__ b0,
                     const float* __restrict__ aff1, const _Float16* __restrict__ w1h,
                     const float* __restrict__ b1, float* __restrict__ stats){
  __shared__ float sfp[96], sw[96], sb[32], sa[64], sb1[32], ls[64];
  int tid = threadIdx.x, wid = tid>>5, lane = tid&31;
  if (tid < 96){ sfp[tid]=fp[tid]; sw[tid]=w0[tid]; }
  if (tid < 64){ sa[tid]=aff1[tid]; ls[tid]=0.f; }
  if (tid < 32){ sb[tid]=b0[tid]; sb1[tid]=b1[tid]; }
  __syncthreads();
  int tile = blockIdx.x*8 + wid;
  int r = tile*16 + (lane & 15);
  v16h a  = frag_a1(r, lane, rel, sfp, sw, sb, sa);
  v16h bf0 = frag_b16(w1h, 32, 0, lane);
  v16h bf1 = frag_b16(w1h, 32, 16, lane);
  v8f c0 = {}, c1 = {};
  c0 = wmma_f16(a, bf0, c0);
  c1 = wmma_f16(a, bf1, c1);
  int cc = lane & 15;
  float s0=0,q0=0,s1=0,q1=0;
  #pragma unroll
  for (int p=0;p<8;p++){
    float v0 = mishf(c0[p] + sb1[cc]);
    float v1 = mishf(c1[p] + sb1[cc+16]);
    s0+=v0; q0+=v0*v0; s1+=v1; q1+=v1*v1;
  }
  s0 += __shfl_xor(s0,16,32); q0 += __shfl_xor(q0,16,32);
  s1 += __shfl_xor(s1,16,32); q1 += __shfl_xor(q1,16,32);
  if (lane < 16){
    atomicAdd(&ls[cc], s0);       atomicAdd(&ls[32+cc], q0);
    atomicAdd(&ls[16+cc], s1);    atomicAdd(&ls[48+cc], q1);
  }
  __syncthreads();
  if (tid < 64) atomicAdd(&stats[tid], ls[tid]);
}

// m1 layer 2 (WMMA) + BN2 affine + layer 3 (32->1) -> h_pre + scalar stats
__global__ void k_h3(const float* __restrict__ rel, const float* __restrict__ fp,
                     const float* __restrict__ w0, const float* __restrict__ b0,
                     const float* __restrict__ aff1, const _Float16* __restrict__ w1h,
                     const float* __restrict__ b1, const float* __restrict__ aff2,
                     const float* __restrict__ w2, const float* __restrict__ b2p,
                     float* __restrict__ hpre, float* __restrict__ stats){
  __shared__ float sfp[96], sw[96], sb[32], sa1[64], sa2[64], sb1[32], sw2[32];
  __shared__ float ta[8][512];
  int tid = threadIdx.x, wid = tid>>5, lane = tid&31;
  if (tid < 96){ sfp[tid]=fp[tid]; sw[tid]=w0[tid]; }
  if (tid < 64){ sa1[tid]=aff1[tid]; sa2[tid]=aff2[tid]; }
  if (tid < 32){ sb[tid]=b0[tid]; sb1[tid]=b1[tid]; sw2[tid]=w2[tid]; }
  float b2 = b2p[0];
  __syncthreads();
  int tile = blockIdx.x*8 + wid;
  int r = tile*16 + (lane & 15);
  v16h a  = frag_a1(r, lane, rel, sfp, sw, sb, sa1);
  v16h bf0 = frag_b16(w1h, 32, 0, lane);
  v16h bf1 = frag_b16(w1h, 32, 16, lane);
  v8f c0 = {}, c1 = {};
  c0 = wmma_f16(a, bf0, c0);
  c1 = wmma_f16(a, bf1, c1);
  int cc = lane & 15, rsel = (lane>=16)?8:0;
  #pragma unroll
  for (int p=0;p<8;p++){
    float v0 = fmaf(mishf(c0[p] + sb1[cc]),    sa2[cc],    sa2[32+cc]);
    float v1 = fmaf(mishf(c1[p] + sb1[cc+16]), sa2[cc+16], sa2[48+cc]);
    ta[wid][(p+rsel)*32 + cc]      = v0;
    ta[wid][(p+rsel)*32 + cc + 16] = v1;
  }
  __syncthreads();
  float hv = 0.f;
  if (lane < 16){
    float acc = b2;
    #pragma unroll
    for (int k=0;k<32;k++) acc = fmaf(ta[wid][lane*32+k], sw2[k], acc);
    hv = mishf(acc);
    hpre[(size_t)tile*16 + lane] = hv;
  }
  float s = wave_sum2(hv);
  float q = wave_sum2(hv*hv);
  if (lane == 0){ atomicAdd(&stats[0], s); atomicAdd(&stats[1], q); }
}

// agg = BN3(h)[16x32] @ w[32x16]  (one WMMA per 16 sl-rows)
__global__ void k_agg(const float* __restrict__ hpre, const float* __restrict__ aff3,
                      const _Float16* __restrict__ wfh, float* __restrict__ wmain){
  int tid = threadIdx.x, wid = tid>>5, lane = tid&31;
  int tile = blockIdx.x*8 + wid;            // < 4096, only i < cM rows
  float s3 = aff3[0], t3 = aff3[1];
  int row = tile*16 + (lane & 15);
  int hs = (lane>=16)?8:0;
  v16h a;
  #pragma unroll
  for (int p=0;p<8;p++){
    int j = ((p<4)?0:16) + hs + ((p&3)<<1);
    a[2*p]   = (_Float16)fmaf(hpre[(size_t)row*32 + j],     s3, t3);
    a[2*p+1] = (_Float16)fmaf(hpre[(size_t)row*32 + j + 1], s3, t3);
  }
  v16h bf = frag_b16(wfh, 16, 0, lane);
  v8f c = {};
  c = wmma_f16(a, bf, c);
  int cc = lane & 15, rsel = (lane>=16)?8:0;
  #pragma unroll
  for (int p=0;p<8;p++)
    wmain[(size_t)(tile*16 + p + rsel)*16 + cc] = c[p];
}

// per-point einsum: f[k,c,d] = sum_m x[b,idx[k,m],c] * wmain[k,m,d]  -> f16 [4096,1024]
__global__ void k_einsum(const float* __restrict__ x, const int* __restrict__ idx,
                         const float* __restrict__ wmain, _Float16* __restrict__ fh){
  __shared__ float swm[8][256];
  int tid = threadIdx.x, wid = tid>>5, lane = tid&31;
  int k = blockIdx.x*8 + wid;
  int b = k >> 9;
  for (int u=lane; u<256; u+=32) swm[wid][u] = wmain[(size_t)k*256 + u];
  __syncthreads();
  const int* myidx = idx + k*16;
  const float* xb = x + (size_t)b*cN*cCIN;
  float acc0[16], acc1[16];
  #pragma unroll
  for (int d=0;d<16;d++){ acc0[d]=0.f; acc1[d]=0.f; }
  for (int m=0;m<16;m++){
    int im = myidx[m];
    float f0 = xb[(size_t)im*64 + lane];
    float f1 = xb[(size_t)im*64 + lane + 32];
    #pragma unroll
    for (int d=0;d<16;d++){
      float w = swm[wid][m*16+d];
      acc0[d] = fmaf(f0, w, acc0[d]);
      acc1[d] = fmaf(f1, w, acc1[d]);
    }
  }
  _Float16* out = fh + (size_t)k*1024;
  #pragma unroll
  for (int d=0;d<16;d++){
    out[lane*16 + d]        = (_Float16)acc0[d];
    out[(lane+32)*16 + d]   = (_Float16)acc1[d];
  }
}

// mr layer 0: [4096,1024] @ [1024,512]
// B tiles double-buffered in LDS via global_load_async_to_lds_b128 (ASYNCcnt)
__global__ void k_mr0(const _Float16* __restrict__ fh, const _Float16* __restrict__ wh,
                      const float* __restrict__ bias, float* __restrict__ r1,
                      float* __restrict__ stats){
  __shared__ _Float16 bt[2][32*16];
  __shared__ float ls[32];
  int tid = threadIdx.x, wid = tid>>5, lane = tid&31;
  int nt = blockIdx.x & 31, mtg = blockIdx.x >> 5;
  int mt = mtg*8 + wid;
  int row = mt*16 + (lane & 15);
  if (tid < 32) ls[tid] = 0.f;
  // prologue: issue async copy of B tile 0 into buffer 0
  if (tid < 64){
    unsigned voff = (unsigned)((((size_t)(tid>>1)*512) + nt*16 + (tid&1)*8) * 2);
    unsigned ldso = (unsigned)(size_t)&bt[0][(tid>>1)*16 + (tid&1)*8];
    asm volatile("global_load_async_to_lds_b128 %0, %1, %2 offset:0"
                 :: "v"(ldso), "v"(voff), "s"(wh) : "memory");
  }
  v8f c = {};
  for (int kk=0; kk<32; ++kk){
    if (tid < 64)
      asm volatile("s_wait_asynccnt 0" ::: "memory");
    __syncthreads();                       // buf[kk&1] ready; prior reads done
    if (kk < 31 && tid < 64){              // overlap next tile copy with compute
      unsigned voff = (unsigned)(((((size_t)(kk+1)*32 + (tid>>1))*512) + nt*16 + (tid&1)*8) * 2);
      unsigned ldso = (unsigned)(size_t)&bt[(kk+1)&1][(tid>>1)*16 + (tid&1)*8];
      asm volatile("global_load_async_to_lds_b128 %0, %1, %2 offset:0"
                   :: "v"(ldso), "v"(voff), "s"(wh) : "memory");
    }
    if (kk < 31)
      __builtin_prefetch(fh + (size_t)row*1024 + (kk+1)*32, 0, 1);
    v16h af = frag_a16(fh, 1024, row, kk*32, lane);
    v16h bf;
    {
      const _Float16* btc = bt[kk&1];
      int nn = lane & 15, kb = (lane>=16)?8:0;
      #pragma unroll
      for (int h=0;h<16;h++) bf[h] = btc[(kb+(h&7)+((h>>3)<<4))*16 + nn];
    }
    c = wmma_f16(af, bf, c);
  }
  int cc = lane & 15, col = nt*16 + cc, rsel = (lane>=16)?8:0;
  float bcol = bias[col], s = 0.f, q = 0.f;
  #pragma unroll
  for (int p=0;p<8;p++){
    float v = mishf(c[p] + bcol);
    r1[(size_t)(mt*16 + p + rsel)*512 + col] = v;
    s += v; q += v*v;
  }
  s += __shfl_xor(s,16,32); q += __shfl_xor(q,16,32);
  if (lane < 16){ atomicAdd(&ls[cc], s); atomicAdd(&ls[16+cc], q); }
  __syncthreads();
  if (tid < 16)       atomicAdd(&stats[nt*16 + tid], ls[tid]);
  else if (tid < 32)  atomicAdd(&stats[512 + nt*16 + tid - 16], ls[tid]);
}

// apply BN0 affine and convert to f16
__global__ void k_affh(const float* __restrict__ in, const float* __restrict__ aff,
                       _Float16* __restrict__ out){
  size_t i = (size_t)blockIdx.x*256 + threadIdx.x;
  int c = (int)(i & 511);
  out[i] = (_Float16)fmaf(in[i], aff[c], aff[512+c]);
}

// mr layer 1: [4096,512] @ [512,64]
__global__ void k_mr1(const _Float16* __restrict__ a, const _Float16* __restrict__ w,
                      const float* __restrict__ bias, float* __restrict__ out,
                      float* __restrict__ stats){
  __shared__ float ls[32];
  int tid = threadIdx.x, wid = tid>>5, lane = tid&31;
  int nt = blockIdx.x & 3, mtg = blockIdx.x >> 2;
  int mt = mtg*8 + wid;
  int row = mt*16 + (lane & 15);
  if (tid < 32) ls[tid] = 0.f;
  __syncthreads();
  v8f c = {};
  for (int kk=0; kk<16; ++kk){
    v16h af = frag_a16(a, 512, row, kk*32, lane);
    v16h bf = frag_b16(w + (size_t)kk*32*64, 64, nt*16, lane);
    c = wmma_f16(af, bf, c);
  }
  int cc = lane & 15, col = nt*16 + cc, rsel = (lane>=16)?8:0;
  float bcol = bias[col], s = 0.f, q = 0.f;
  #pragma unroll
  for (int p=0;p<8;p++){
    float v = mishf(c[p] + bcol);
    out[(size_t)(mt*16 + p + rsel)*64 + col] = v;
    s += v; q += v*v;
  }
  s += __shfl_xor(s,16,32); q += __shfl_xor(q,16,32);
  if (lane < 16){ atomicAdd(&ls[cc], s); atomicAdd(&ls[16+cc], q); }
  __syncthreads();
  if (tid < 16)       atomicAdd(&stats[nt*16 + tid], ls[tid]);
  else if (tid < 32)  atomicAdd(&stats[64 + nt*16 + tid - 16], ls[tid]);
}

__global__ void k_out(const float* __restrict__ in, const float* __restrict__ aff,
                      float* __restrict__ out){
  size_t i = (size_t)blockIdx.x*256 + threadIdx.x;
  int c = (int)(i & 63);
  out[i] = fmaf(in[i], aff[c], aff[64+c]);
}

// ---------------- launch ----------------
extern "C" void kernel_launch(void* const* d_in, const int* in_sizes, int n_in,
                              void* d_out, int out_size, void* d_ws, size_t ws_size,
                              hipStream_t stream) {
  const float* x    = (const float*)d_in[0];
  const float* xyz  = (const float*)d_in[1];
  const float* fpts = (const float*)d_in[2];
  const float* fpw  = (const float*)d_in[3];
  const float* m1w0=(const float*)d_in[4],  *m1b0=(const float*)d_in[5],
             *m1g0=(const float*)d_in[6],  *m1be0=(const float*)d_in[7];
  const float* m1w1=(const float*)d_in[8],  *m1b1=(const float*)d_in[9],
             *m1g1=(const float*)d_in[10], *m1be1=(const float*)d_in[11];
  const float* m1w2=(const float*)d_in[12], *m1b2=(const float*)d_in[13],
             *m1g2=(const float*)d_in[14], *m1be2=(const float*)d_in[15];
  const float* m2w0=(const float*)d_in[16], *m2b0=(const float*)d_in[17],
             *m2g0=(const float*)d_in[18], *m2be0=(const float*)d_in[19];
  const float* m2w1=(const float*)d_in[20], *m2b1=(const float*)d_in[21],
             *m2g1=(const float*)d_in[22], *m2be1=(const float*)d_in[23];
  const float* mrw0=(const float*)d_in[24], *mrb0=(const float*)d_in[25],
             *mrg0=(const float*)d_in[26], *mrbe0=(const float*)d_in[27];
  const float* mrw1=(const float*)d_in[28], *mrb1=(const float*)d_in[29],
             *mrg1=(const float*)d_in[30], *mrbe1=(const float*)d_in[31];
  (void)in_sizes; (void)n_in; (void)out_size; (void)ws_size;

  char* ws = (char*)d_ws;
  float*     w_fp    = (float*)(ws + OFF_FP);
  int*       w_idx   = (int*)(ws + OFF_IDX);
  float*     w_rel   = (float*)(ws + OFF_REL);
  float*     w_st    = (float*)(ws + OFF_STATS);
  _Float16*  w_w1h   = (_Float16*)(ws + OFF_W1H);
  _Float16*  w_wfh   = (_Float16*)(ws + OFF_WFH);
  _Float16*  w_mrw1h = (_Float16*)(ws + OFF_MRW1H);
  _Float16*  w_mrw0h = (_Float16*)(ws + OFF_MRW0H);
  float*     w_m2t1  = (float*)(ws + OFF_M2T1);
  float*     w_hpre  = (float*)(ws + OFF_HPRE);
  float*     w_wmain = (float*)(ws + OFF_WMAIN);
  _Float16*  w_fh    = (_Float16*)(ws + OFF_FH);
  float*     w_r1    = (float*)(ws + OFF_R1);
  _Float16*  w_r1h   = (_Float16*)(ws + OFF_R1H);
  float*     w_r2    = (float*)(ws + OFF_R2);

  hipMemsetAsync(w_st, 0, 8192*sizeof(float), stream);

  k_prep    <<<1,               256, 0, stream>>>(fpts, m1w1, mrw1, w_fp, w_w1h, w_mrw1h);
  k_conv_f2h<<<524288/256,      256, 0, stream>>>(mrw0, w_mrw0h, 524288);
  k_knn     <<<cBN/256,         256, 0, stream>>>(x, w_idx);
  k_rel     <<<cM/256,          256, 0, stream>>>(xyz, w_idx, w_rel);
  k_m2a     <<<64,              256, 0, stream>>>(fpw, m2w0, m2b0, w_m2t1, w_st + ST_M2B0);
  k_m2b     <<<1,               512, 0, stream>>>(w_m2t1, w_st + ST_M2B0, m2g0, m2be0,
                                                  m2w1, m2b1, m2g1, m2be1, w_wfh);
  k_h1      <<<cRT/256,         256, 0, stream>>>(w_rel, w_fp, m1w0, m1b0, w_st + ST_BN1);
  k_bnfin   <<<1,               32,  0, stream>>>(w_st + ST_BN1, w_st + ST_BN1A,
                                                  m1g0, m1be0, (float)cRT, 32);
  k_h2      <<<cRT/(16*8),      256, 0, stream>>>(w_rel, w_fp, m1w0, m1b0, w_st + ST_BN1A,
                                                  w_w1h, m1b1, w_st + ST_BN2);
  k_bnfin   <<<1,               32,  0, stream>>>(w_st + ST_BN2, w_st + ST_BN2A,
                                                  m1g1, m1be1, (float)cRT, 32);
  k_h3      <<<cRT/(16*8),      256, 0, stream>>>(w_rel, w_fp, m1w0, m1b0, w_st + ST_BN1A,
                                                  w_w1h, m1b1, w_st + ST_BN2A,
                                                  m1w2, m1b2, w_hpre, w_st + ST_BN3);
  k_bnfin   <<<1,               32,  0, stream>>>(w_st + ST_BN3, w_st + ST_BN3A,
                                                  m1g2, m1be2, (float)cRT, 1);
  k_agg     <<<cM/(16*8),       256, 0, stream>>>(w_hpre, w_st + ST_BN3A, w_wfh, w_wmain);
  k_einsum  <<<cBN/8,           256, 0, stream>>>(x, w_idx, w_wmain, w_fh);
  k_mr0     <<<1024,            256, 0, stream>>>(w_fh, w_mrw0h, mrb0, w_r1, w_st + ST_MRB0);
  k_bnfin   <<<2,               256, 0, stream>>>(w_st + ST_MRB0, w_st + ST_MRB0A,
                                                  mrg0, mrbe0, 4096.f, 512);
  k_affh    <<<2097152/256,     256, 0, stream>>>(w_r1, w_st + ST_MRB0A, w_r1h);
  k_mr1     <<<128,             256, 0, stream>>>(w_r1h, w_mrw1h, mrb1, w_r2, w_st + ST_MRB1);
  k_bnfin   <<<1,               64,  0, stream>>>(w_st + ST_MRB1, w_st + ST_MRB1A,
                                                  mrg1, mrbe1, 4096.f, 64);
  k_out     <<<262144/256,      256, 0, stream>>>(w_r2, w_st + ST_MRB1A, (float*)d_out);
}
